// MambaBlock_34883724378128
// MI455X (gfx1250) — compile-verified, matrix operand order checked by reading, not run on hardware
//
#include <hip/hip_runtime.h>
#include <hip/hip_bf16.h>
#include <math.h>

// ---------------- problem constants ----------------
constexpr int D_MODEL   = 1024;
constexpr int D_STATE   = 16;
constexpr int D_CONV    = 4;
constexpr int HEADDIM   = 8;
constexpr int D_INNER   = 2048;             // EXPAND * D_MODEL
constexpr int NHEADS    = 256;              // D_INNER / HEADDIM
constexpr int CONV_DIM  = D_INNER + 2 * D_STATE;          // 2080
constexpr int D_IN_PROJ = 2 * D_INNER + 2 * D_STATE + NHEADS; // 4384
constexpr int B_SZ      = 4;
constexpr int SEQ       = 2048;
constexpr int ROWS      = B_SZ * SEQ;       // 8192
constexpr float EPS     = 1e-5f;

// ---------------- vector types ----------------
typedef __attribute__((ext_vector_type(16))) __bf16 v16bf;
typedef __attribute__((ext_vector_type(8)))  __bf16 v8bf;
typedef __attribute__((ext_vector_type(8)))  float  v8f;

// 128-bit payload type + AS-qualified pointers for the async LDS builtins
typedef int b128i __attribute__((vector_size(16)));
typedef __attribute__((address_space(1))) b128i* b128i_gptr; // global
typedef __attribute__((address_space(3))) b128i* b128i_lptr; // LDS

// CDNA5 async global->LDS path (probe-guarded; falls back to sync staging)
#if __has_builtin(__builtin_amdgcn_global_load_async_to_lds_b128) && \
    __has_builtin(__builtin_amdgcn_s_wait_asynccnt)
#define USE_ASYNC_LDS 1
#else
#define USE_ASYNC_LDS 0
#endif

__device__ __forceinline__ float silu_f(float x) {
    return x * (1.0f / (1.0f + __expf(-x)));
}
__device__ __forceinline__ float softplus_f(float x) {
    return (x > 20.0f) ? x : log1pf(__expf(x));
}

// ============================================================
// Kernel 1: LayerNorm over D_MODEL, emit bf16 activations (GEMM A)
// ============================================================
__global__ __launch_bounds__(256)
void k_layernorm_cast(const float* __restrict__ x,
                      const float* __restrict__ ln_w,
                      const float* __restrict__ ln_b,
                      __bf16* __restrict__ xn_bf)
{
    const int row = blockIdx.x;
    const int tid = threadIdx.x;
    const float* xr = x + (size_t)row * D_MODEL;

    float v[4];
    float s = 0.f, ss = 0.f;
#pragma unroll
    for (int i = 0; i < 4; ++i) {
        v[i] = xr[tid + i * 256];
        s  += v[i];
        ss += v[i] * v[i];
    }
    __shared__ float red[2][256];
    red[0][tid] = s; red[1][tid] = ss;
    __syncthreads();
    for (int off = 128; off > 0; off >>= 1) {
        if (tid < off) {
            red[0][tid] += red[0][tid + off];
            red[1][tid] += red[1][tid + off];
        }
        __syncthreads();
    }
    const float mean = red[0][0] * (1.0f / D_MODEL);
    const float var  = red[1][0] * (1.0f / D_MODEL) - mean * mean;
    const float rstd = rsqrtf(var + EPS);

    __bf16* outr = xn_bf + (size_t)row * D_MODEL;
#pragma unroll
    for (int i = 0; i < 4; ++i) {
        const int c = tid + i * 256;
        const float y = (v[i] - mean) * rstd * ln_w[c] + ln_b[c];
        outr[c] = (__bf16)y;
    }
}

// ============================================================
// Kernel 2: transpose + cast f32 (K x N) -> bf16 (N x K)
// ============================================================
__global__ void k_transpose_cast(const float* __restrict__ in,
                                 __bf16* __restrict__ out,
                                 int K, int N)
{
    const size_t idx = (size_t)blockIdx.x * blockDim.x + threadIdx.x;
    const size_t total = (size_t)K * N;
    if (idx >= total) return;
    const int k = (int)(idx / N);
    const int n = (int)(idx % N);
    out[(size_t)n * K + k] = (__bf16)in[idx];
}

// ============================================================
// Kernel 3: WMMA bf16 GEMM  C[M,N] = A[M,K] * B[K,N] (+resid)
// Block tile 128x128, 8 waves (2M x 4N), wave tile 64x32,
// per-wave 4x2 accumulators of v_wmma_f32_16x16x32_bf16.
// Double-buffered LDS; async global->LDS staging when available
// (ASYNCcnt + one barrier per K-step), sync fallback otherwise.
// ============================================================
#define TM 128
#define TN 128
#define TK 32

__global__ __launch_bounds__(256)
void k_gemm_bf16_wmma(const __bf16* __restrict__ A,
                      const __bf16* __restrict__ Bt,
                      float* __restrict__ C,
                      const float* __restrict__ resid, // may be null
                      int M, int N, int K)
{
    __shared__ __align__(16) __bf16 As[2][TM][TK]; // 2 x 8 KB
    __shared__ __align__(16) __bf16 Bs[2][TN][TK]; // 2 x 8 KB

    const int tid  = threadIdx.x;
    const int wave = tid >> 5;
    const int lane = tid & 31;
    const int block_m = blockIdx.y * TM;
    const int block_n = blockIdx.x * TN;

    const int wm = (wave & 1) * 64;   // wave M offset in tile
    const int wn = (wave >> 1) * 32;  // wave N offset in tile

    const int lrow  = lane & 15;      // fragment row/col index
    const int khalf = lane >> 4;      // 0 -> K{0..7,16..23}, 1 -> K{8..15,24..31}

    v8f acc[4][2] = {};

    // staging coordinates: 4 threads x 16B per 64B row
    const int ldr = tid >> 2;          // 0..63
    const int ldc = (tid & 3) * 8;     // half offset, 16B aligned

    const bool fullN = (block_n + TN) <= N;

    auto stage = [&](int kt, int buf) {
        const int k0 = kt * TK;
#pragma unroll
        for (int pass = 0; pass < 2; ++pass) {
            const int r = ldr + pass * 64;
            const __bf16* srcA = A + (size_t)(block_m + r) * K + k0 + ldc;
#if USE_ASYNC_LDS
            __builtin_amdgcn_global_load_async_to_lds_b128(
                (b128i_gptr)srcA,
                (b128i_lptr)&As[buf][r][ldc],
                0, 0);
#else
            *(v8bf*)(&As[buf][r][ldc]) = *(const v8bf*)srcA;
#endif
            const int gn = block_n + r;
            const __bf16* srcB = Bt + (size_t)gn * K + k0 + ldc;
            if (fullN) {
#if USE_ASYNC_LDS
                __builtin_amdgcn_global_load_async_to_lds_b128(
                    (b128i_gptr)srcB,
                    (b128i_lptr)&Bs[buf][r][ldc],
                    0, 0);
#else
                *(v8bf*)(&Bs[buf][r][ldc]) = *(const v8bf*)srcB;
#endif
            } else {
                v8bf val = {};
                if (gn < N) val = *(const v8bf*)srcB;
                *(v8bf*)(&Bs[buf][r][ldc]) = val;
            }
        }
    };

    const int ksteps = K / TK;
    stage(0, 0);

    for (int kt = 0; kt < ksteps; ++kt) {
        const int cur = kt & 1;
#if USE_ASYNC_LDS
        __builtin_amdgcn_s_wait_asynccnt(0);
#endif
        __syncthreads();
        // stage next tile into the other buffer while computing this one
        if (kt + 1 < ksteps) stage(kt + 1, cur ^ 1);

        // ---- B fragments (2) ----
        v16bf bfr[2];
#pragma unroll
        for (int ni = 0; ni < 2; ++ni) {
            const int col = wn + ni * 16 + lrow;
            v8bf blo = *(const v8bf*)(&Bs[cur][col][khalf * 8]);
            v8bf bhi = *(const v8bf*)(&Bs[cur][col][16 + khalf * 8]);
            bfr[ni] = __builtin_shufflevector(blo, bhi,
                0,1,2,3,4,5,6,7,8,9,10,11,12,13,14,15);
        }
        // ---- A fragments (4) + 8 WMMAs ----
#pragma unroll
        for (int mi = 0; mi < 4; ++mi) {
            const int rowi = wm + mi * 16 + lrow;
            v8bf alo = *(const v8bf*)(&As[cur][rowi][khalf * 8]);
            v8bf ahi = *(const v8bf*)(&As[cur][rowi][16 + khalf * 8]);
            v16bf afr = __builtin_shufflevector(alo, ahi,
                0,1,2,3,4,5,6,7,8,9,10,11,12,13,14,15);
#pragma unroll
            for (int ni = 0; ni < 2; ++ni) {
                acc[mi][ni] = __builtin_amdgcn_wmma_f32_16x16x32_bf16(
                    false, afr, false, bfr[ni], (short)0, acc[mi][ni],
                    false, false);
            }
        }
    }

    // ---- epilogue: C/D layout (ISA 7.12.2): vgpr g, lanes0-15 -> M=g,
    // lanes16-31 -> M=8+g; N = lane&15 ----
    const int m_half = (lane < 16) ? 0 : 8;
    const int nn     = lane & 15;
#pragma unroll
    for (int mi = 0; mi < 4; ++mi) {
#pragma unroll
        for (int ni = 0; ni < 2; ++ni) {
            const int colg = block_n + wn + ni * 16 + nn;
            if (colg >= N) continue;
#pragma unroll
            for (int g = 0; g < 8; ++g) {
                const int rowg = block_m + wm + mi * 16 + m_half + g;
                float v = acc[mi][ni][g];
                if (resid) v += resid[(size_t)rowg * N + colg];
                C[(size_t)rowg * N + colg] = v;
            }
        }
    }
}

// ============================================================
// Kernel 4: depthwise causal conv (k=4) + SiLU over xBC channels
// ============================================================
__global__ void k_conv_silu(const float* __restrict__ u,
                            const float* __restrict__ conv_w,
                            const float* __restrict__ conv_b,
                            float* __restrict__ xconv)
{
    const size_t idx = (size_t)blockIdx.x * blockDim.x + threadIdx.x;
    const size_t total = (size_t)ROWS * CONV_DIM;
    if (idx >= total) return;
    const int row = (int)(idx / CONV_DIM);
    const int c   = (int)(idx % CONV_DIM);
    const int b   = row / SEQ;
    const int l   = row % SEQ;

    float accv = conv_b[c];
#pragma unroll
    for (int k = 0; k < D_CONV; ++k) {
        const int ls = l - (D_CONV - 1) + k;
        if (ls >= 0) {
            accv += conv_w[k * CONV_DIM + c] *
                    u[((size_t)b * SEQ + ls) * D_IN_PROJ + D_INNER + c];
        }
    }
    xconv[idx] = silu_f(accv);
}

// ============================================================
// Kernel 5: dt = softplus(u[..., -NHEADS:] + dt_bias)
// ============================================================
__global__ void k_dt(const float* __restrict__ u,
                     const float* __restrict__ dt_bias,
                     float* __restrict__ dtb)
{
    const size_t idx = (size_t)blockIdx.x * blockDim.x + threadIdx.x;
    if (idx >= (size_t)ROWS * NHEADS) return;
    const int row = (int)(idx / NHEADS);
    const int h   = (int)(idx % NHEADS);
    const float v = u[(size_t)row * D_IN_PROJ + (D_IN_PROJ - NHEADS) + h] + dt_bias[h];
    dtb[idx] = softplus_f(v);
}

// ============================================================
// Kernel 6: selective scan, chunked through LDS.
// One block (128 threads) per (b, head); thread (p = tid>>4, n = tid&15)
// owns state element h[p][n]. 32 timesteps staged per chunk so global
// traffic is bulk/coalesced instead of serial-latency bound; recurrence
// runs out of LDS; reduction over n via shfl_xor in 16-lane groups.
// ============================================================
constexpr int SCAN_CH = 32;

__global__ __launch_bounds__(128)
void k_scan(const float* __restrict__ xconv,
            const float* __restrict__ dtb,
            const float* __restrict__ A_log,
            const float* __restrict__ Dp,
            float* __restrict__ yb)
{
    const int bh = blockIdx.x;
    const int b  = bh / NHEADS;
    const int h  = bh % NHEADS;
    const int tid = threadIdx.x;
    const int n = tid & 15;     // state dim
    const int p = tid >> 4;     // head dim

    const float Ah  = -__expf(A_log[h]);
    const float Dph = Dp[h];

    __shared__ float sx[SCAN_CH][HEADDIM];
    __shared__ float sB[SCAN_CH][D_STATE];
    __shared__ float sC[SCAN_CH][D_STATE];
    __shared__ float sdt[SCAN_CH];
    __shared__ float sy[SCAN_CH][HEADDIM];

    float hs = 0.0f;
    for (int c0 = 0; c0 < SEQ; c0 += SCAN_CH) {
        // ---- cooperative stage of 32 timesteps ----
        for (int i = tid; i < SCAN_CH * HEADDIM; i += 128) {
            const int l = i >> 3, pp = i & 7;
            sx[l][pp] = xconv[((size_t)b * SEQ + c0 + l) * CONV_DIM + h * HEADDIM + pp];
        }
        for (int i = tid; i < SCAN_CH * D_STATE; i += 128) {
            const int l = i >> 4, nn = i & 15;
            const size_t base = ((size_t)b * SEQ + c0 + l) * CONV_DIM + D_INNER;
            sB[l][nn] = xconv[base + nn];
            sC[l][nn] = xconv[base + D_STATE + nn];
        }
        for (int i = tid; i < SCAN_CH; i += 128)
            sdt[i] = dtb[((size_t)b * SEQ + c0 + i) * NHEADS + h];
        __syncthreads();

        // ---- recurrence over the chunk (LDS + ALU only) ----
        for (int l = 0; l < SCAN_CH; ++l) {
            const float dt = sdt[l];
            const float xv = sx[l][p];
            const float dA = __expf(dt * Ah);
            hs = hs * dA + dt * xv * sB[l][n];
            float yp = hs * sC[l][n];
            yp += __shfl_xor(yp, 8, 16);
            yp += __shfl_xor(yp, 4, 16);
            yp += __shfl_xor(yp, 2, 16);
            yp += __shfl_xor(yp, 1, 16);
            if (n == 0) sy[l][p] = yp + Dph * xv;
        }
        __syncthreads();

        // ---- coalesced writeback ----
        for (int i = tid; i < SCAN_CH * HEADDIM; i += 128) {
            const int l = i >> 3, pp = i & 7;
            yb[((size_t)b * SEQ + c0 + l) * D_INNER + h * HEADDIM + pp] = sy[l][pp];
        }
        __syncthreads();
    }
}

// ============================================================
// Kernel 7: g = y * silu(z); RMS-norm over D_INNER; emit bf16
// ============================================================
__global__ __launch_bounds__(256)
void k_gate_rms_cast(const float* __restrict__ yb,
                     const float* __restrict__ u,
                     const float* __restrict__ rms_w,
                     __bf16* __restrict__ gb)
{
    const int row = blockIdx.x;
    const int tid = threadIdx.x;
    const float* yr = yb + (size_t)row * D_INNER;
    const float* zr = u  + (size_t)row * D_IN_PROJ; // z = first D_INNER cols

    float g[8];
    float ss = 0.f;
#pragma unroll
    for (int i = 0; i < 8; ++i) {
        const int c = tid + i * 256;
        const float gv = yr[c] * silu_f(zr[c]);
        g[i] = gv;
        ss += gv * gv;
    }
    __shared__ float red[256];
    red[tid] = ss;
    __syncthreads();
    for (int off = 128; off > 0; off >>= 1) {
        if (tid < off) red[tid] += red[tid + off];
        __syncthreads();
    }
    const float rms = rsqrtf(red[0] * (1.0f / D_INNER) + EPS);

    __bf16* outr = gb + (size_t)row * D_INNER;
#pragma unroll
    for (int i = 0; i < 8; ++i) {
        const int c = tid + i * 256;
        outr[c] = (__bf16)(g[i] * rms * rms_w[c]);
    }
}

// ============================================================
// launch
// ============================================================
extern "C" void kernel_launch(void* const* d_in, const int* in_sizes, int n_in,
                              void* d_out, int out_size, void* d_ws, size_t ws_size,
                              hipStream_t stream)
{
    const float* x       = (const float*)d_in[0];
    const float* ln_w    = (const float*)d_in[1];
    const float* ln_b    = (const float*)d_in[2];
    const float* W_in    = (const float*)d_in[3];
    const float* conv_w  = (const float*)d_in[4];
    const float* conv_b  = (const float*)d_in[5];
    const float* A_log   = (const float*)d_in[6];
    const float* Dp      = (const float*)d_in[7];
    const float* dt_bias = (const float*)d_in[8];
    const float* rms_w   = (const float*)d_in[9];
    const float* W_out   = (const float*)d_in[10];
    float* out = (float*)d_out;

    // ---- workspace carve-up (256B aligned) ----
    char* ws = (char*)d_ws;
    size_t off = 0;
    auto carve = [&](size_t bytes) -> char* {
        char* p = ws + off;
        off = (off + bytes + 255) & ~(size_t)255;
        return p;
    };
    __bf16* xn_bf  = (__bf16*)carve((size_t)ROWS * D_MODEL * 2);
    __bf16* WtIn   = (__bf16*)carve((size_t)D_IN_PROJ * D_MODEL * 2);
    float*  u      = (float*) carve((size_t)ROWS * D_IN_PROJ * 4);
    float*  xconv  = (float*) carve((size_t)ROWS * CONV_DIM * 4);
    float*  dtb    = (float*) carve((size_t)ROWS * NHEADS * 4);
    float*  yb     = (float*) carve((size_t)ROWS * D_INNER * 4);
    __bf16* gb     = (__bf16*)carve((size_t)ROWS * D_INNER * 2);
    __bf16* WtOut  = (__bf16*)carve((size_t)D_MODEL * D_INNER * 2);
    (void)ws_size; (void)in_sizes; (void)n_in; (void)out_size;

    // 1. layernorm + cast
    k_layernorm_cast<<<ROWS, 256, 0, stream>>>(x, ln_w, ln_b, xn_bf);

    // 2. W_in transpose+cast  (K=D_MODEL, N=D_IN_PROJ)
    {
        const size_t tot = (size_t)D_MODEL * D_IN_PROJ;
        k_transpose_cast<<<(int)((tot + 255) / 256), 256, 0, stream>>>(
            W_in, WtIn, D_MODEL, D_IN_PROJ);
    }

    // 3. u = xn @ W_in   (8192 x 4384 x 1024)
    {
        dim3 grid((D_IN_PROJ + TN - 1) / TN, ROWS / TM);
        k_gemm_bf16_wmma<<<grid, 256, 0, stream>>>(
            xn_bf, WtIn, u, nullptr, ROWS, D_IN_PROJ, D_MODEL);
    }

    // 4. conv + silu
    {
        const size_t tot = (size_t)ROWS * CONV_DIM;
        k_conv_silu<<<(int)((tot + 255) / 256), 256, 0, stream>>>(
            u, conv_w, conv_b, xconv);
    }

    // 5. dt
    {
        const size_t tot = (size_t)ROWS * NHEADS;
        k_dt<<<(int)((tot + 255) / 256), 256, 0, stream>>>(u, dt_bias, dtb);
    }

    // 6. scan
    k_scan<<<B_SZ * NHEADS, 128, 0, stream>>>(xconv, dtb, A_log, Dp, yb);

    // 7. gate + RMS + cast
    k_gate_rms_cast<<<ROWS, 256, 0, stream>>>(yb, u, rms_w, gb);

    // 8. W_out transpose+cast (K=D_INNER, N=D_MODEL)
    {
        const size_t tot = (size_t)D_INNER * D_MODEL;
        k_transpose_cast<<<(int)((tot + 255) / 256), 256, 0, stream>>>(
            W_out, WtOut, D_INNER, D_MODEL);
    }

    // 9. out = gb @ W_out + x   (8192 x 1024 x 2048)
    {
        dim3 grid(D_MODEL / TN, ROWS / TM);
        k_gemm_bf16_wmma<<<grid, 256, 0, stream>>>(
            gb, WtOut, out, x, ROWS, D_MODEL, D_INNER);
    }
}
